// MicrostateDGFLayer_14328010899510
// MI455X (gfx1250) — matrix operand order repaired and checked by reading, not compiled, not run on hardware
//
#include <hip/hip_runtime.h>

#define BB 8
#define NN 2048
#define DD 128
#define HH 8

constexpr float LN_EPS = 1e-5f;
constexpr float KERN_EPS = 1e-6f;

typedef __attribute__((ext_vector_type(16))) __bf16 v16bf;
typedef __attribute__((ext_vector_type(8)))  float v8f;

union FragBF { v16bf v; unsigned int u[8]; };

__device__ __forceinline__ unsigned short f2bf(float f) {
  unsigned int b = __float_as_uint(f);
  unsigned int r = b + 0x7FFFu + ((b >> 16) & 1u);   // round-to-nearest-even
  return (unsigned short)(r >> 16);
}
__device__ __forceinline__ unsigned int pack2bf(float lo, float hi) {
  return (unsigned int)f2bf(lo) | ((unsigned int)f2bf(hi) << 16);
}
// A-matrix (16x32 bf16) k index for vgpr-pair v, lane-half hi  (ISA 7.12.2)
__device__ __forceinline__ int ka_map(int v, int hi) {
  return ((v < 4) ? 2 * v : 16 + 2 * (v - 4)) + (hi ? 8 : 0);
}
// B-matrix (32x16 bf16) k index, same order as A (ka_map)
__device__ __forceinline__ int kb_map(int v, int hi) {
  return ((v < 4) ? 2 * v : 16 + 2 * (v - 4)) + (hi ? 8 : 0);
}

// ---------------- Kernel 0: row sumsq + bf16 row-major + bf16 col-major ----
__global__ __launch_bounds__(128) void prep_kernel(
    const float* __restrict__ x, float* __restrict__ sq,
    unsigned short* __restrict__ xbf, unsigned short* __restrict__ xT) {
  const int row = blockIdx.x;              // b*N + n
  const int b = row / NN, n = row % NN;
  const int d = threadIdx.x;
  float v = x[(size_t)row * DD + d];
  xbf[(size_t)row * DD + d] = f2bf(v);
  xT[((size_t)b * DD + d) * NN + n] = f2bf(v);
  __shared__ float red[128];
  red[d] = v * v;
  __syncthreads();
  for (int s = 64; s > 0; s >>= 1) {
    if (d < s) red[d] += red[d + s];
    __syncthreads();
  }
  if (d == 0) sq[row] = red[0];
}

// ---------------- Kernel 1: Gram (WMMA bf16) -> dist -> multihead exp -> gate
__global__ __launch_bounds__(256) void adj_kernel(
    const unsigned short* __restrict__ xbf, const float* __restrict__ sq,
    const float* __restrict__ prev_adj, const float* __restrict__ log_sigmas,
    const float* __restrict__ gate_in, float* __restrict__ adj_out) {
  const int b = blockIdx.z;
  const int i_base = blockIdx.y * 128;
  const int j_base = blockIdx.x * 128;
  const int wave = threadIdx.x >> 5;
  const int lane = threadIdx.x & 31;
  const int hi = lane >> 4;
  const int lnm = lane & 15;

  float invden[HH];
#pragma unroll
  for (int h = 0; h < HH; ++h) {
    float s = __expf(log_sigmas[h]);
    invden[h] = 1.0f / (2.0f * s * s + KERN_EPS);
  }
  const float g = 1.0f / (1.0f + __expf(-gate_in[0]));

  const int i0 = i_base + wave * 16;
  FragBF afr[4];
  {
    const unsigned int* rp =
        (const unsigned int*)xbf + (size_t)(b * NN + i0 + lnm) * (DD / 2);
#pragma unroll
    for (int kc = 0; kc < 4; ++kc)
#pragma unroll
      for (int v = 0; v < 8; ++v)
        afr[kc].u[v] = rp[(kc * 32 + ka_map(v, hi)) >> 1];
  }
  float sqi[8];
#pragma unroll
  for (int v = 0; v < 8; ++v) sqi[v] = sq[b * NN + i0 + v + 8 * hi];

  for (int jt = 0; jt < 8; ++jt) {
    const int j0 = j_base + jt * 16;

    // prefetch next j-tile's prev_adj rows (largest HBM stream)
    if (jt < 7) {
#pragma unroll
      for (int v = 0; v < 8; ++v) {
        const size_t pidx =
            ((size_t)b * NN + (i0 + v + 8 * hi)) * NN + (j0 + 16 + lnm);
        __builtin_prefetch(&prev_adj[pidx], 0, 3);
      }
    }

    FragBF bfr[4];
    const unsigned int* cp =
        (const unsigned int*)xbf + (size_t)(b * NN + j0 + lnm) * (DD / 2);
#pragma unroll
    for (int kc = 0; kc < 4; ++kc)
#pragma unroll
      for (int v = 0; v < 8; ++v)
        bfr[kc].u[v] = cp[(kc * 32 + kb_map(v, hi)) >> 1];
    v8f c = {};
#pragma unroll
    for (int kc = 0; kc < 4; ++kc)
      c = __builtin_amdgcn_wmma_f32_16x16x32_bf16(
          false, afr[kc].v, false, bfr[kc].v, (short)0, c, false, false);

    const float sqj = sq[b * NN + j0 + lnm];
#pragma unroll
    for (int v = 0; v < 8; ++v) {
      const int gi = i0 + v + 8 * hi;
      const int gj = j0 + lnm;
      float dist = fmaxf(sqi[v] + sqj - 2.0f * c[v], 0.0f);
      float acc = 0.0f;
#pragma unroll
      for (int h = 0; h < HH; ++h) acc += __expf(-dist * invden[h]);
      acc *= (1.0f / HH);
      const size_t idx = ((size_t)b * NN + gi) * NN + gj;
      adj_out[idx] = g * acc + (1.0f - g) * prev_adj[idx];
    }
  }
}

// ---------------- Kernel 2: h = adj@x (K=2048), proj = h@W^T, ELU+res+LN ----
#define AP 17     // s_a row pitch (uints)  -> conflict-free 16-lane column read
#define HP 130    // s_h row pitch (floats) -> conflict-free 16-lane column read

__global__ __launch_bounds__(256) void prop_kernel(
    const float* __restrict__ x, const unsigned short* __restrict__ xT,
    const float* __restrict__ adj, const float* __restrict__ W,
    const float* __restrict__ bias, const float* __restrict__ gamma,
    const float* __restrict__ beta, float* __restrict__ out) {
  const int b = blockIdx.y;
  const int i0 = blockIdx.x * 16;
  const int t = threadIdx.x;
  const int wave = t >> 5, lane = t & 31, hi = lane >> 4, lnm = lane & 15;

  __shared__ unsigned int s_a[2][16 * AP];  // double-buffered 16x32 bf16 chunk
  __shared__ float s_h[16 * HP];            // h tile / activation tile
  __shared__ float s_red[2 * 16 * 16];
  __shared__ float s_mu[16], s_rs[16];

  // ---- GEMM 1: 16 rows x 128 cols, K = 2048, double-buffered staging ----
  v8f c = {};
  const int dcol = wave * 16 + lnm;                 // this lane's output column d
  const unsigned int* xtp =
      (const unsigned int*)xT + (size_t)(b * DD + dcol) * (NN / 2);
  const int srow = t >> 4;                          // staging row 0..15
  const int spair = t & 15;                         // staging pair 0..15
  const float* aprow =
      adj + ((size_t)b * NN + (i0 + srow)) * NN + spair * 2;

  // stage chunk 0
  s_a[0][srow * AP + spair] = pack2bf(aprow[0], aprow[1]);
  __syncthreads();

  for (int kc = 0; kc < NN / 32; ++kc) {
    const int buf = kc & 1;
    FragBF af, bf_;
#pragma unroll
    for (int v = 0; v < 8; ++v) {
      af.u[v] = s_a[buf][lnm * AP + (ka_map(v, hi) >> 1)];
      bf_.u[v] = xtp[(kc * 32 + kb_map(v, hi)) >> 1];
    }
    c = __builtin_amdgcn_wmma_f32_16x16x32_bf16(
        false, af.v, false, bf_.v, (short)0, c, false, false);
    if (kc + 1 < NN / 32) {                         // stage next chunk
      const float* ap = aprow + (kc + 1) * 32;
      s_a[buf ^ 1][srow * AP + spair] = pack2bf(ap[0], ap[1]);
    }
    __syncthreads();
  }
#pragma unroll
  for (int v = 0; v < 8; ++v) s_h[(v + 8 * hi) * HP + dcol] = c[v];
  __syncthreads();

  // ---- GEMM 2: proj = h @ W^T  (K = 128) ----
  v8f c2 = {};
#pragma unroll
  for (int kc = 0; kc < 4; ++kc) {
    FragBF af, bf_;
#pragma unroll
    for (int v = 0; v < 8; ++v) {
      const int k0 = kc * 32 + ka_map(v, hi);
      af.u[v] = pack2bf(s_h[lnm * HP + k0], s_h[lnm * HP + k0 + 1]);
      const int k0b = kc * 32 + kb_map(v, hi);
      const float* wp = W + (size_t)(wave * 16 + lnm) * DD + k0b;
      bf_.u[v] = pack2bf(wp[0], wp[1]);
    }
    c2 = __builtin_amdgcn_wmma_f32_16x16x32_bf16(
        false, af.v, false, bf_.v, (short)0, c2, false, false);
  }

  // ---- bias + ELU + residual ----
  const float bval = bias[dcol];
  __syncthreads();                                   // done reading s_h
#pragma unroll
  for (int v = 0; v < 8; ++v) {
    const int m = v + 8 * hi;
    float hv = c2[v] + bval;
    float e = hv > 0.0f ? hv : (__expf(hv) - 1.0f);
    float a = e + x[((size_t)b * NN + i0 + m) * DD + dcol];
    s_h[m * HP + dcol] = a;
  }
  __syncthreads();

  // ---- LayerNorm over D=128 ----
  {
    float sm = 0.f, s2 = 0.f;
#pragma unroll
    for (int cix = spair * 8; cix < spair * 8 + 8; ++cix) {
      float a = s_h[srow * HP + cix];
      sm += a;
      s2 += a * a;
    }
    s_red[srow * 16 + spair] = sm;
    s_red[256 + srow * 16 + spair] = s2;
  }
  __syncthreads();
  if (t < 16) {
    float sm = 0.f, s2 = 0.f;
#pragma unroll
    for (int jx = 0; jx < 16; ++jx) {
      sm += s_red[t * 16 + jx];
      s2 += s_red[256 + t * 16 + jx];
    }
    float mu = sm * (1.0f / DD);
    float var = s2 * (1.0f / DD) - mu * mu;
    s_mu[t] = mu;
    s_rs[t] = rsqrtf(var + LN_EPS);
  }
  __syncthreads();
#pragma unroll
  for (int cix = spair * 8; cix < spair * 8 + 8; ++cix) {
    float a = s_h[srow * HP + cix];
    float o = (a - s_mu[srow]) * s_rs[srow] * gamma[cix] + beta[cix];
    out[((size_t)b * NN + i0 + srow) * DD + cix] = o;
  }
}

extern "C" void kernel_launch(void* const* d_in, const int* in_sizes, int n_in,
                              void* d_out, int out_size, void* d_ws,
                              size_t ws_size, hipStream_t stream) {
  const float* x          = (const float*)d_in[0];
  const float* prev_adj   = (const float*)d_in[1];
  const float* log_sigmas = (const float*)d_in[2];
  const float* tg         = (const float*)d_in[3];
  const float* W          = (const float*)d_in[4];
  const float* bias       = (const float*)d_in[5];
  const float* gamma      = (const float*)d_in[6];
  const float* beta       = (const float*)d_in[7];

  float* out     = (float*)d_out;
  float* adj_out = out + (size_t)BB * NN * DD;       // tuple: out, adj_final

  char* ws = (char*)d_ws;
  float* sq            = (float*)ws;                                    // B*N f32
  unsigned short* xbf  = (unsigned short*)(ws + (size_t)BB * NN * 4);   // B*N*D bf16
  unsigned short* xT   = xbf + (size_t)BB * NN * DD;                    // B*D*N bf16

  prep_kernel<<<BB * NN, 128, 0, stream>>>(x, sq, xbf, xT);

  dim3 g1(NN / 128, NN / 128, BB);
  adj_kernel<<<g1, 256, 0, stream>>>(xbf, sq, prev_adj, log_sigmas, tg, adj_out);

  dim3 g2(NN / 16, BB, 1);
  prop_kernel<<<g2, 256, 0, stream>>>(x, xT, adj_out, W, bias, gamma, beta, out);
}
